// MLA_20212116095645
// MI455X (gfx1250) — matrix operand order repaired and checked by reading, not compiled, not run on hardware
//
#include <hip/hip_runtime.h>

// ---------------- problem constants ----------------
#define BSZ   2
#define SEQ   2048
#define DIM   2048
#define NH    16
#define QLORA 1536
#define KVLORA 512
#define ROPE_D 64
#define QKHD  192      // NOPE(128)+ROPE(64)
#define VHD   128
#define TOK   (BSZ*SEQ)   // 4096
#define QSCALE 0.07216878364870323f   // 1/sqrt(192)

typedef __bf16 bf16;
typedef __bf16 v16bf __attribute__((ext_vector_type(16)));
typedef __bf16 v8bf  __attribute__((ext_vector_type(8)));
typedef float  v8f   __attribute__((ext_vector_type(8)));
typedef int    v4i   __attribute__((ext_vector_type(4)));

#define DEV static __device__ __forceinline__

// ---------------- CDNA5 async copy (global -> LDS), guarded ----------------
#if defined(__has_builtin)
#if __has_builtin(__builtin_amdgcn_global_load_async_to_lds_b128)
#define USE_ASYNC 1
#endif
#endif
#ifndef USE_ASYNC
#define USE_ASYNC 0
#endif

DEV void async_copy16(const bf16* g, bf16* l) {
#if USE_ASYNC
  // prototype (from hipcc diagnostic): (int4 AS1*, int4 AS3*, imm offset, imm cpol)
  __builtin_amdgcn_global_load_async_to_lds_b128(
      (__attribute__((address_space(1))) v4i*)g,
      (__attribute__((address_space(3))) v4i*)l, 0, 0);
#else
  *reinterpret_cast<v8bf*>(l) = *reinterpret_cast<const v8bf*>(g);
#endif
}

template <int N>
DEV void wait_asynccnt() {
#if USE_ASYNC
#if __has_builtin(__builtin_amdgcn_s_wait_asynccnt)
  __builtin_amdgcn_s_wait_asynccnt(N);
#else
  asm volatile("s_wait_asynccnt %0" ::"i"(N));
#endif
#endif
}

// ---------------- WMMA helpers (CDNA5 bf16 16x16x32, f32 accum) -------------
DEV v8f wmma_bf16(v16bf a, v16bf b, v8f c) {
  // 8 args: (neg_a, A, neg_b, B, c_mod, C, reuse_a, reuse_b)
  return __builtin_amdgcn_wmma_f32_16x16x32_bf16(false, a, false, b, (short)0, c,
                                                 false, false);
}

// A fragment (16x32, 16-bit): lane holds row (lane&15); two contiguous 16B
// chunks at K = hi*8 and K = 16+hi*8 per ISA layout.
DEV v16bf load_frag_a(const bf16* row, int k0, int hi) {
  v8bf lo = *reinterpret_cast<const v8bf*>(row + k0 + hi * 8);
  v8bf hv = *reinterpret_cast<const v8bf*>(row + k0 + 16 + hi * 8);
  v16bf r;
#pragma unroll
  for (int j = 0; j < 8; ++j) { r[j] = lo[j]; r[j + 8] = hv[j]; }
  return r;
}

// B fragment (32x16, 16-bit): lane holds column (lane&15), two 16B chunks at
// K = hi*8 and K = 16+hi*8 (column n of B == row n of W when B = W^T).
DEV v16bf load_frag_b(const bf16* row, int k0, int hi) {
  return load_frag_a(row, k0, hi);
}

// 16-lane row reductions (C/D layout: row m = r + 8*(lane>>4), col n = lane&15)
DEV float red_max16(float v) {
  v = fmaxf(v, __shfl_xor(v, 1));
  v = fmaxf(v, __shfl_xor(v, 2));
  v = fmaxf(v, __shfl_xor(v, 4));
  v = fmaxf(v, __shfl_xor(v, 8));
  return v;
}
DEV float red_sum16(float v) {
  v += __shfl_xor(v, 1);
  v += __shfl_xor(v, 2);
  v += __shfl_xor(v, 4);
  v += __shfl_xor(v, 8);
  return v;
}

// ---------------- f32 -> bf16 conversion ----------------
__global__ void cvt_f32_bf16(const float* __restrict__ in, bf16* __restrict__ out,
                             size_t n) {
  size_t i = (size_t)blockIdx.x * blockDim.x + threadIdx.x;
  size_t stride = (size_t)gridDim.x * blockDim.x;
  for (; i < n; i += stride) out[i] = (bf16)in[i];
}

// ---------------- WMMA GEMM: C[M,N] = A[M,K] * W[N,K]^T + bias --------------
// Block: 256 threads = 8 waves (4 M x 2 N), 128x128 block tile, K-chunks of 32
// double-buffered in LDS via async-to-LDS copies. LDS pitch 40 (bank-padded).
// W buffer must be allocated with ceil(N/128)*128 rows (stores masked to N).
#define GP 40   // LDS pitch for 32-wide K chunks

template <int OUT_F32>
__global__ __launch_bounds__(256) void gemm_bt_wmma(
    const bf16* __restrict__ A, const bf16* __restrict__ W,
    float* __restrict__ Cf, bf16* __restrict__ Cb,
    const float* __restrict__ bias,
    int M, int N, int K, int lda, int ldb, int ldc) {
  __shared__ bf16 As[2][128 * GP];
  __shared__ bf16 Bs[2][128 * GP];

  int tid = threadIdx.x;
  int bm = blockIdx.x % (M >> 7);
  int bn = blockIdx.x / (M >> 7);
  int w = tid >> 5;
  int wm = w & 3, wn = w >> 2;
  unsigned lane = tid & 31u;
  int n16 = lane & 15, hi = lane >> 4;

  const bf16* Ag = A + (size_t)bm * 128 * lda;
  const bf16* Wg = W + (size_t)bn * 128 * ldb;

  // stage one 128x32 A chunk + 128x32 B chunk: 2+2 b128 copies per thread
  auto stage = [&](int buf, int k0) {
#pragma unroll
    for (int i = 0; i < 2; ++i) {
      int L = tid + 256 * i;           // 0..511
      int row = L >> 2, c8 = (L & 3) * 8;
      async_copy16(Ag + (size_t)row * lda + k0 + c8, &As[buf][row * GP + c8]);
      async_copy16(Wg + (size_t)row * ldb + k0 + c8, &Bs[buf][row * GP + c8]);
    }
  };

  v8f acc[2][4];
#pragma unroll
  for (int i = 0; i < 2; ++i)
#pragma unroll
    for (int j = 0; j < 4; ++j) acc[i][j] = (v8f){0, 0, 0, 0, 0, 0, 0, 0};

  int kt_total = K >> 5;
  stage(0, 0);
  for (int kt = 0; kt < kt_total; ++kt) {
    int cur = kt & 1;
    if (kt + 1 < kt_total) {
      stage(cur ^ 1, (kt + 1) * 32);
      wait_asynccnt<4>();              // previous chunk's 4 copies landed
    } else {
      wait_asynccnt<0>();
    }
    __syncthreads();

    const bf16* as = &As[cur][0];
    const bf16* bs = &Bs[cur][0];
    v16bf a0 = load_frag_a(as + (wm * 32 + n16) * GP, 0, hi);
    v16bf a1 = load_frag_a(as + (wm * 32 + 16 + n16) * GP, 0, hi);
#pragma unroll
    for (int j = 0; j < 4; ++j) {
      v16bf b = load_frag_b(bs + (wn * 64 + j * 16 + n16) * GP, 0, hi);
      acc[0][j] = wmma_bf16(a0, b, acc[0][j]);
      acc[1][j] = wmma_bf16(a1, b, acc[1][j]);
    }
    __syncthreads();
  }

#pragma unroll
  for (int i = 0; i < 2; ++i)
#pragma unroll
    for (int j = 0; j < 4; ++j) {
      int col = bn * 128 + wn * 64 + j * 16 + n16;
      if (col < N) {
        float bv = bias ? bias[col] : 0.f;
#pragma unroll
        for (int r = 0; r < 8; ++r) {
          int row = bm * 128 + wm * 32 + i * 16 + hi * 8 + r;
          float v = acc[i][j][r] + bv;
          if (OUT_F32)
            Cf[(size_t)row * ldc + col] = v;
          else
            Cb[(size_t)row * ldc + col] = (bf16)v;
        }
      }
    }
}

// ---------------- RMSNorm: row-wise, f32 in -> bf16 out ----------------
__global__ __launch_bounds__(256) void rmsnorm_kernel(
    const float* __restrict__ in, const float* __restrict__ w,
    bf16* __restrict__ out, int D, int inStride, int outStride) {
  __shared__ float red[256];
  int row = blockIdx.x;
  const float* xr = in + (size_t)row * inStride;
  float s = 0.f;
  for (int i = threadIdx.x; i < D; i += 256) { float v = xr[i]; s += v * v; }
  red[threadIdx.x] = s;
  __syncthreads();
  for (int st = 128; st > 0; st >>= 1) {
    if ((int)threadIdx.x < st) red[threadIdx.x] += red[threadIdx.x + st];
    __syncthreads();
  }
  float rms = rsqrtf(red[0] / (float)D + 1e-6f);
  bf16* orow = out + (size_t)row * outStride;
  for (int i = threadIdx.x; i < D; i += 256)
    orow[i] = (bf16)(xr[i] * rms * w[i]);
}

// ---------------- assemble q_full / k_full / V^T (+RoPE, +scale) -------------
__global__ __launch_bounds__(256) void assemble_kernel(
    const bf16* __restrict__ q, const bf16* __restrict__ kvb,
    const float* __restrict__ kv, bf16* __restrict__ qfull,
    bf16* __restrict__ kfull, bf16* __restrict__ vt) {
  int idx = blockIdx.x * blockDim.x + threadIdx.x;   // over B*H*SEQ
  if (idx >= BSZ * NH * SEQ) return;
  int s = idx % SEQ;
  int h = (idx / SEQ) % NH;
  int b = idx / (SEQ * NH);
  size_t bs = (size_t)b * SEQ + s;
  const bf16* qrow = q + bs * (NH * QKHD) + h * QKHD;
  const bf16* kvbrow = kvb + bs * (NH * 256) + h * 256;
  const float* kpe = kv + bs * (KVLORA + ROPE_D) + KVLORA;
  bf16* qf = qfull + ((size_t)(b * NH + h) * SEQ + s) * QKHD;
  bf16* kf = kfull + ((size_t)(b * NH + h) * SEQ + s) * QKHD;
  bf16* vtb = vt + (size_t)(b * NH + h) * VHD * SEQ;

  for (int d = 0; d < 128; ++d) qf[d] = (bf16)((float)qrow[d] * QSCALE);
  for (int d = 0; d < 128; ++d) kf[d] = kvbrow[d];

  float t = (float)s;
  for (int d = 0; d < ROPE_D; ++d) {
    int i = d & 31;
    float invf = __expf(-((float)i / 32.f) * 9.210340371976184f);  // 10000^(-2i/64)
    float ang = t * invf;
    float c = __cosf(ang), sn = __sinf(ang);
    float qv = (float)qrow[128 + d];
    float qr = (d < 32) ? -(float)qrow[128 + d + 32] : (float)qrow[128 + d - 32];
    qf[128 + d] = (bf16)((qv * c + qr * sn) * QSCALE);
    float kvv = kpe[d];
    float kr = (d < 32) ? -kpe[d + 32] : kpe[d - 32];
    kf[128 + d] = (bf16)(kvv * c + kr * sn);
  }
  for (int d = 0; d < VHD; ++d) vtb[(size_t)d * SEQ + s] = kvbrow[128 + d];
}

// ---------------- flash attention ----------------
// Block = 128 threads = 4 waves, all for one (b,h); wave w owns queries
// qBase + w*16 .. +16. K/V^T chunks (32 keys) are async-staged into
// double-buffered LDS and shared by the 4 waves (4x traffic reduction).
#define KP 200   // LDS pitch of a K row (192 elems, bank-padded)

__global__ __launch_bounds__(128) void attn_kernel(
    const bf16* __restrict__ qfull, const bf16* __restrict__ kfull,
    const bf16* __restrict__ vt, bf16* __restrict__ attn_out) {
  __shared__ bf16 Ks[2][32 * KP];
  __shared__ bf16 Vs[2][128 * GP];
  __shared__ bf16 Pb[4][16 * 32];

  int tid = threadIdx.x;
  int w = tid >> 5;
  const int QB = SEQ / 64;                 // 32 query-blocks per (b,h)
  int qb = blockIdx.x % QB;
  int h = (blockIdx.x / QB) % NH;
  int b = blockIdx.x / (QB * NH);
  unsigned lane = tid & 31u;
  int n = lane & 15, hi = lane >> 4;
  int qBase = qb * 64;
  int qRow = qBase + w * 16;               // this wave's query tile

  const bf16* Q = qfull + ((size_t)(b * NH + h) * SEQ + qRow) * QKHD;
  const bf16* Kg = kfull + (size_t)(b * NH + h) * SEQ * QKHD;
  const bf16* Vg = vt + (size_t)(b * NH + h) * VHD * SEQ;
  bf16* P = &Pb[w][0];

  // stage one 32-key chunk of K (32x192) and V^T (128x32): 6+4 copies/thread
  auto stage = [&](int buf, int k0) {
#pragma unroll
    for (int i = 0; i < 6; ++i) {
      int L = tid + 128 * i;               // 0..767
      int row = L / 24, c8 = (L % 24) * 8;
      async_copy16(Kg + (size_t)(k0 + row) * QKHD + c8, &Ks[buf][row * KP + c8]);
    }
#pragma unroll
    for (int i = 0; i < 4; ++i) {
      int L = tid + 128 * i;               // 0..511
      int row = L >> 2, c8 = (L & 3) * 8;
      async_copy16(Vg + (size_t)row * SEQ + k0 + c8, &Vs[buf][row * GP + c8]);
    }
  };

  // Q fragments live in VGPRs across the whole key loop (6 x 16x32)
  v16bf qf[6];
  const bf16* qrow = Q + (size_t)n * QKHD;
#pragma unroll
  for (int d = 0; d < 6; ++d) qf[d] = load_frag_a(qrow, d * 32, hi);

  v8f o[8];
#pragma unroll
  for (int t = 0; t < 8; ++t) o[t] = (v8f){0, 0, 0, 0, 0, 0, 0, 0};
  float rmax[8], rsum[8];
#pragma unroll
  for (int r = 0; r < 8; ++r) { rmax[r] = -3.0e38f; rsum[r] = 0.f; }

  int totalKeys = qRow + 16;               // this wave's causal horizon
  int blockKeys = qBase + 64;              // block max horizon
  stage(0, 0);
  for (int k0 = 0; k0 < blockKeys; k0 += 32) {
    int cur = (k0 >> 5) & 1;
    if (k0 + 32 < blockKeys) {
      stage(cur ^ 1, k0 + 32);
      wait_asynccnt<10>();
    } else {
      wait_asynccnt<0>();
    }
    __syncthreads();

    if (k0 < totalKeys) {                  // wave-uniform: skip past-horizon
      const bf16* ks = &Ks[cur][0];
      const bf16* vs = &Vs[cur][0];
      bool hasB = (k0 + 16) < totalKeys;   // wave-uniform
      v8f sa = (v8f){0, 0, 0, 0, 0, 0, 0, 0};
      v8f sb = (v8f){0, 0, 0, 0, 0, 0, 0, 0};
#pragma unroll
      for (int d = 0; d < 6; ++d) {
        v16bf kfa = load_frag_b(ks + n * KP, d * 32, hi);
        sa = wmma_bf16(qf[d], kfa, sa);
      }
      if (hasB) {
#pragma unroll
        for (int d = 0; d < 6; ++d) {
          v16bf kfb = load_frag_b(ks + (16 + n) * KP, d * 32, hi);
          sb = wmma_bf16(qf[d], kfb, sb);
        }
      }
      // online softmax (scores pre-scaled via Q); causal mask per element
#pragma unroll
      for (int r = 0; r < 8; ++r) {
        int m = r + hi * 8;
        int qg = qRow + m;
        float va = sa[r];
        if (k0 + n > qg) va = -3.0e38f;
        float vb = -3.0e38f;
        if (hasB) { vb = sb[r]; if (k0 + 16 + n > qg) vb = -3.0e38f; }
        float tm = red_max16(fmaxf(va, vb));
        float nm = fmaxf(rmax[r], tm);
        float corr = __expf(rmax[r] - nm);
        rmax[r] = nm;
        float pa = __expf(va - nm);
        float pb = hasB ? __expf(vb - nm) : 0.f;
        rsum[r] = rsum[r] * corr + red_sum16(pa + pb);
#pragma unroll
        for (int t = 0; t < 8; ++t) o[t][r] *= corr;
        P[m * 32 + n] = (bf16)pa;
        P[m * 32 + 16 + n] = (bf16)pb;
      }
      asm volatile("s_wait_dscnt 0x0" ::: "memory");   // P store -> load fence
      v16bf pf = load_frag_a(P + n * 32, 0, hi);
#pragma unroll
      for (int t = 0; t < 8; ++t) {
        v16bf vf = load_frag_b(vs + (t * 16 + n) * GP, 0, hi);
        o[t] = wmma_bf16(pf, vf, o[t]);
      }
    }
    __syncthreads();
  }

  bf16* outb = attn_out + ((size_t)b * SEQ + qRow) * (NH * VHD) + h * VHD;
#pragma unroll
  for (int r = 0; r < 8; ++r) {
    int m = r + hi * 8;
    float inv = 1.f / rsum[r];
#pragma unroll
    for (int t = 0; t < 8; ++t)
      outb[(size_t)m * (NH * VHD) + t * 16 + n] = (bf16)(o[t][r] * inv);
  }
}

// ---------------- launcher ----------------
extern "C" void kernel_launch(void* const* d_in, const int* in_sizes, int n_in,
                              void* d_out, int out_size, void* d_ws, size_t ws_size,
                              hipStream_t stream) {
  (void)in_sizes; (void)n_in; (void)out_size; (void)ws_size;
  const float* x       = (const float*)d_in[0];
  const float* wq_a_w  = (const float*)d_in[1];
  const float* wq_a_b  = (const float*)d_in[2];
  const float* q_norm  = (const float*)d_in[3];
  const float* wq_b_w  = (const float*)d_in[4];
  const float* wq_b_b  = (const float*)d_in[5];
  const float* wkv_a_w = (const float*)d_in[6];
  const float* wkv_a_b = (const float*)d_in[7];
  const float* kv_norm = (const float*)d_in[8];
  const float* wkv_b_w = (const float*)d_in[9];
  const float* wkv_b_b = (const float*)d_in[10];
  const float* wo_w    = (const float*)d_in[11];
  const float* wo_b    = (const float*)d_in[12];
  float* out = (float*)d_out;

  const size_t n_x    = (size_t)TOK * DIM;
  const size_t n_wqa  = (size_t)QLORA * DIM;
  const size_t n_wqb  = (size_t)NH * QKHD * QLORA;
  const size_t n_wkva = (size_t)(KVLORA + ROPE_D) * DIM;       // 576 rows
  const size_t n_wkva_pad = (size_t)640 * DIM;                 // pad to 5*128
  const size_t n_wkvb = (size_t)NH * 256 * KVLORA;
  const size_t n_wo   = (size_t)DIM * NH * VHD;

  char* ws = (char*)d_ws;
  size_t off = 0;
  auto alloc = [&](size_t bytes) {
    size_t r = off;
    off += (bytes + 255) & ~(size_t)255;
    return r;
  };
  bf16* xb    = (bf16*)(ws + alloc(n_x * 2));
  bf16* wqa   = (bf16*)(ws + alloc(n_wqa * 2));
  bf16* wqb   = (bf16*)(ws + alloc(n_wqb * 2));
  bf16* wkva  = (bf16*)(ws + alloc(n_wkva_pad * 2));
  bf16* wkvb  = (bf16*)(ws + alloc(n_wkvb * 2));
  bf16* wob   = (bf16*)(ws + alloc(n_wo * 2));
  float* qa   = (float*)(ws + alloc((size_t)TOK * QLORA * 4));
  bf16* qn    = (bf16*)(ws + alloc((size_t)TOK * QLORA * 2));
  bf16* qb    = (bf16*)(ws + alloc((size_t)TOK * NH * QKHD * 2));
  float* kvf  = (float*)(ws + alloc((size_t)TOK * (KVLORA + ROPE_D) * 4));
  bf16* kvn   = (bf16*)(ws + alloc((size_t)TOK * KVLORA * 2));
  bf16* kvbb  = (bf16*)(ws + alloc((size_t)TOK * NH * 256 * 2));
  bf16* qfull = (bf16*)(ws + alloc((size_t)BSZ * NH * SEQ * QKHD * 2));
  bf16* kfull = (bf16*)(ws + alloc((size_t)BSZ * NH * SEQ * QKHD * 2));
  bf16* vt    = (bf16*)(ws + alloc((size_t)BSZ * NH * VHD * SEQ * 2));
  bf16* aout  = (bf16*)(ws + alloc((size_t)TOK * NH * VHD * 2));

  // 1) convert inputs/weights to bf16 once
  cvt_f32_bf16<<<2048, 256, 0, stream>>>(x, xb, n_x);
  cvt_f32_bf16<<<1024, 256, 0, stream>>>(wq_a_w, wqa, n_wqa);
  cvt_f32_bf16<<<1024, 256, 0, stream>>>(wq_b_w, wqb, n_wqb);
  cvt_f32_bf16<<<1024, 256, 0, stream>>>(wkv_a_w, wkva, n_wkva);
  cvt_f32_bf16<<<1024, 256, 0, stream>>>(wkv_b_w, wkvb, n_wkvb);
  cvt_f32_bf16<<<1024, 256, 0, stream>>>(wo_w, wob, n_wo);

  auto gemm_grid = [](int M, int N) { return (M / 128) * ((N + 127) / 128); };

  // 2) q_a = x @ wq_a^T + b  (f32 out for rmsnorm)
  gemm_bt_wmma<1><<<gemm_grid(TOK, QLORA), 256, 0, stream>>>(
      xb, wqa, qa, nullptr, wq_a_b, TOK, QLORA, DIM, DIM, DIM, QLORA);
  // 3) rmsnorm(q_a) -> bf16
  rmsnorm_kernel<<<TOK, 256, 0, stream>>>(qa, q_norm, qn, QLORA, QLORA, QLORA);
  // 4) q = qn @ wq_b^T + b  (bf16 out)
  gemm_bt_wmma<0><<<gemm_grid(TOK, NH * QKHD), 256, 0, stream>>>(
      qn, wqb, nullptr, qb, wq_b_b, TOK, NH * QKHD, QLORA, QLORA, QLORA, NH * QKHD);
  // 5) kv = x @ wkv_a^T + b  (f32 out: rmsnorm + rope need it; N padded to 640)
  gemm_bt_wmma<1><<<gemm_grid(TOK, KVLORA + ROPE_D), 256, 0, stream>>>(
      xb, wkva, kvf, nullptr, wkv_a_b, TOK, KVLORA + ROPE_D, DIM, DIM, DIM,
      KVLORA + ROPE_D);
  // 6) rmsnorm(kv_c) -> bf16
  rmsnorm_kernel<<<TOK, 256, 0, stream>>>(kvf, kv_norm, kvn, KVLORA,
                                          KVLORA + ROPE_D, KVLORA);
  // 7) kvb = kvn @ wkv_b^T + b (bf16 out)
  gemm_bt_wmma<0><<<gemm_grid(TOK, NH * 256), 256, 0, stream>>>(
      kvn, wkvb, nullptr, kvbb, wkv_b_b, TOK, NH * 256, KVLORA, KVLORA, KVLORA,
      NH * 256);
  // 8) assemble q_full (RoPE+scale), k_full (RoPE on k_pe broadcast), V^T
  assemble_kernel<<<(BSZ * NH * SEQ + 255) / 256, 256, 0, stream>>>(
      qb, kvbb, kvf, qfull, kfull, vt);
  // 9) flash attention (WMMA QK^T + WMMA PV, block-shared async K/V tiles)
  attn_kernel<<<BSZ * NH * (SEQ / 64), 128, 0, stream>>>(qfull, kfull, vt, aout);
  // 10) out = aout @ wo^T + b  (f32 to d_out)
  gemm_bt_wmma<1><<<gemm_grid(TOK, DIM), 256, 0, stream>>>(
      aout, wob, out, nullptr, wo_b, TOK, DIM, NH * VHD, NH * VHD, NH * VHD, DIM);
}
